// LSTMClassifier_58128087384177
// MI455X (gfx1250) — matrix operand order repaired and checked by reading, not compiled, not run on hardware
//
#include <hip/hip_runtime.h>
#include <hip/hip_bf16.h>

// ---------------------------------------------------------------------------
// LSTM classifier for MI455X (gfx1250, wave32, WMMA + TDM).
//   K0 embcvt:      emb f32 [50000,300] -> f16 padded [50000,320]
//   K1 prep:        swizzle w_ih / w_hh into WMMA B-fragment order (f16), bias sum
//   K2 proj GEMM:   proj[50000,512] = emb x w_ih^T + bias (f16 WMMA, 16x64/wave)
//   K3 recurrence:  persistent kernel, 64 blocks x 16 batch rows, 200 steps,
//                   W_hh resident in LDS, xg gathered by the Tensor Data Mover
//                   (gather-mode D#, double-buffered, TENSORcnt-synced),
//                   gates via WMMA f16 -> f32
//   K4 fc head:     128 -> 64 relu -> 1 sigmoid
// ---------------------------------------------------------------------------

typedef __attribute__((ext_vector_type(16))) _Float16 v16h;
typedef __attribute__((ext_vector_type(8)))  float    v8f;
typedef __attribute__((ext_vector_type(4)))  unsigned uint4v;
typedef __attribute__((ext_vector_type(4)))  unsigned u32x4;
typedef __attribute__((ext_vector_type(8)))  int      i32x8;
typedef __attribute__((ext_vector_type(4)))  int      i32x4;

union FragH {             // one WMMA 16-bit A/B operand per lane (8 VGPRs)
    v16h    h;
    unsigned u[8];
    uint4v   q[2];
};

__device__ __forceinline__ unsigned pack2h(float a, float b) {
    union { _Float16 h[2]; unsigned u; } t;
    t.h[0] = (_Float16)a; t.h[1] = (_Float16)b;
    return t.u;
}

__device__ __forceinline__ float sigmoidf_(float x) {
    return 1.0f / (1.0f + __expf(-x));
}
__device__ __forceinline__ float tanh_fast(float x) {
    // tanh(x) = 1 - 2/(exp(2x)+1)
    return 1.0f - 2.0f / (__expf(2.0f * x) + 1.0f);
}

#define VOCAB 50000
#define EMB   300
#define EMBP  320          // K padded to 10*32
#define HID   128
#define G4    512          // 4*HID
#define BB    1024
#define TT    200
#define NT_W  32           // 512/16 N tiles
#define KT_IH 10           // 320/32
#define KT_HH 4            // 128/32

#define N_WIH (KT_IH * NT_W * 32 * 16)   // 163840 swizzled halves
#define N_WHH (KT_HH * NT_W * 32 * 16)   // 65536 swizzled halves

// K3 dynamic LDS layout (bytes)
#define LDS_WHH   0          // 131072 B : W_hh B-fragments
#define LDS_H     131072     //   4096 B : h state, f16 16x128
#define LDS_G     135168     //  32768 B : gates,  f32 16x512
#define LDS_XG    167936     //  32768 B : 2 x (16 x 512 f16) xg double buffer
#define LDS_TOTAL 200704

// A-fragment per-lane base half-offset inside a 16x32 k-tile:
//   q[0] = halves K = kt*32 + (lane<16?0:8) + 0..7
//   q[1] = halves K = kt*32 + (lane<16?0:8) + 16..23
__device__ __forceinline__ int a_base_k(int kt, int lane) {
    return kt * 32 + ((lane & 16) ? 8 : 0);
}

// ---------------------------------------------------------------------------
// K0: embedding f32 -> f16, rows padded 300 -> 320 with zeros.
// ---------------------------------------------------------------------------
__global__ void embcvt_kernel(const float* __restrict__ emb, _Float16* __restrict__ emb_h) {
    int i = blockIdx.x * blockDim.x + threadIdx.x;      // 0 .. 50000*160-1
    if (i >= VOCAB * (EMBP / 2)) return;
    int row = i / (EMBP / 2);
    int kp  = (i - row * (EMBP / 2)) * 2;
    unsigned packed = 0u;
    if (kp < EMB) {
        const float2* p = (const float2*)(emb + (size_t)row * EMB + kp);
        packed = pack2h(p->x, p->y);
    }
    *(unsigned*)(emb_h + (size_t)row * EMBP + kp) = packed;
}

// ---------------------------------------------------------------------------
// K1: weight swizzle into B-fragment order + bias sum
// Storage: flat i = ((kt*32+nt)*32 + lane)*16 + h  holds  B[k][n] with
//   k = kt*32 + (lane<16 ? 0:8) + (h&7) + (h<8 ? 0:16) ,  n = nt*16 + (lane&15)
// ---------------------------------------------------------------------------
__global__ void prep_kernel(const float* __restrict__ w_ih, const float* __restrict__ w_hh,
                            const float* __restrict__ b_ih, const float* __restrict__ b_hh,
                            _Float16* __restrict__ wih_swz, _Float16* __restrict__ whh_swz,
                            float* __restrict__ bsum) {
    int i = blockIdx.x * blockDim.x + threadIdx.x;
    if (i < N_WIH) {
        int hh = i & 15, ln = (i >> 4) & 31, tt = i >> 9;
        int nt = tt & 31, kt = tt >> 5;
        int k = kt * 32 + ((ln & 16) ? 8 : 0) + (hh & 7) + ((hh & 8) ? 16 : 0);
        int n = nt * 16 + (ln & 15);
        wih_swz[i] = (k < EMB) ? (_Float16)w_ih[n * EMB + k] : (_Float16)0.f;
    } else if (i < N_WIH + N_WHH) {
        int ii = i - N_WIH;
        int hh = ii & 15, ln = (ii >> 4) & 31, tt = ii >> 9;
        int nt = tt & 31, kt = tt >> 5;
        int k = kt * 32 + ((ln & 16) ? 8 : 0) + (hh & 7) + ((hh & 8) ? 16 : 0);
        int n = nt * 16 + (ln & 15);
        whh_swz[ii] = (_Float16)w_hh[n * HID + k];
    } else if (i < N_WIH + N_WHH + G4) {
        int j = i - N_WIH - N_WHH;
        bsum[j] = b_ih[j] + b_hh[j];
    }
}

// ---------------------------------------------------------------------------
// K2: proj[v][g] = sum_e emb[v][e] * w_ih[g][e] + bsum[g], stored f16.
// One wave per 16x64 output strip (A-fragment reused over 4 N tiles).
// ---------------------------------------------------------------------------
__global__ void proj_gemm_kernel(const _Float16* __restrict__ emb_h,
                                 const _Float16* __restrict__ wih_swz,
                                 const float* __restrict__ bsum,
                                 _Float16* __restrict__ proj) {
    const int lane  = threadIdx.x & 31;
    const int wave  = blockIdx.x * 8 + (threadIdx.x >> 5);
    const int mtile = wave >> 3;            // 0..3124
    const int ng    = wave & 7;             // strip: ntiles 4*ng .. 4*ng+3
    const int m     = mtile * 16 + (lane & 15);
    const int rofs  = (lane & 16) ? 8 : 0;

    v8f acc[4];
#pragma unroll
    for (int s = 0; s < 4; ++s) acc[s] = (v8f){0.f, 0.f, 0.f, 0.f, 0.f, 0.f, 0.f, 0.f};

    const _Float16* arow = emb_h + (size_t)m * EMBP;
    for (int kt = 0; kt < KT_IH; ++kt) {
        FragH fa;
        const uint4v* ap = (const uint4v*)(arow + a_base_k(kt, lane));
        fa.q[0] = ap[0];            // halves +0..7
        fa.q[1] = ap[2];            // halves +16..23
#pragma unroll
        for (int s = 0; s < 4; ++s) {
            const int nt = ng * 4 + s;
            FragH fb;
            const uint4v* bp = (const uint4v*)(wih_swz + ((size_t)(kt * 32 + nt) * 32 + lane) * 16);
            fb.q[0] = bp[0];
            fb.q[1] = bp[1];
            acc[s] = __builtin_amdgcn_wmma_f32_16x16x32_f16(false, fa.h, false, fb.h,
                                                            (short)0, acc[s], false, false);
        }
    }
#pragma unroll
    for (int s = 0; s < 4; ++s) {
        const int col = (ng * 4 + s) * 16 + (lane & 15);
        float bs = bsum[col];
#pragma unroll
        for (int j = 0; j < 8; ++j) {
            int row = mtile * 16 + rofs + j;
            proj[(size_t)row * G4 + col] = (_Float16)(acc[s][j] + bs);
        }
    }
}

// ---------------------------------------------------------------------------
// K3: persistent recurrence. Block = 256 threads (8 waves), owns 16 batch rows.
// xg rows are DMA-gathered into LDS by the Tensor Data Mover (gather-mode D#,
// 16 x 16-bit row indices, tile 16x512 f16), double buffered across steps.
// ---------------------------------------------------------------------------
__global__ void lstm_recurrence_kernel(const int* __restrict__ x,
                                       const _Float16* __restrict__ proj,
                                       const _Float16* __restrict__ whh_swz,
                                       float* __restrict__ hn) {
    extern __shared__ char smem[];
    _Float16* s_whh = (_Float16*)(smem + LDS_WHH);
    _Float16* s_h   = (_Float16*)(smem + LDS_H);
    float*    s_g   = (float*)(smem + LDS_G);
    _Float16* s_xg  = (_Float16*)(smem + LDS_XG);       // 2 buffers of 16x512

    const int tid   = threadIdx.x;
    const int lane  = tid & 31;
    const int wv    = tid >> 5;
    const int btile = blockIdx.x;
    const int rofs  = (lane & 16) ? 8 : 0;
    const int mrow  = lane & 15;

    // Raw LDS byte offset of the xg buffers (flat LDS address truncates to offset).
    const unsigned xg_off = (unsigned)(size_t)(smem + LDS_XG);
    const unsigned long long ga = (unsigned long long)(size_t)proj;

    // TDM gather-mode issue: pull 16 rows (x[b, t]) of proj[*,512] f16 into LDS.
    auto tdm_issue = [&](int t) {
        unsigned idx[16];
#pragma unroll
        for (int r = 0; r < 16; ++r)
            idx[r] = (unsigned)x[(btile * 16 + r) * TT + t] & 0xFFFFu;
        u32x4 g0;
        g0.x = 0x80000001u;                              // count=1 | gather_mode, 16b idx
        g0.y = xg_off + ((t & 1) ? 16384u : 0u);         // lds_addr (bytes)
        g0.z = (unsigned)ga;                             // global_addr[31:0]
        g0.w = ((unsigned)(ga >> 32) & 0x01FFFFFFu) | 0x80000000u; // ga[56:32] | type=2
        i32x8 g1;
        g1[0] = 0x00010000;                              // data_size=1 (2B), mask=0
        g1[1] = (int)(512u << 16);                       // tensor_dim0 = 512
        g1[2] = (int)((unsigned)VOCAB << 16);            // tensor_dim1 lo
        g1[3] = (int)(512u << 16);                       // tensor_dim1 hi | tile_dim0=512
        g1[4] = 16;                                      // tile_dim1 = #indices
        g1[5] = 512;                                     // tensor_dim0_stride = 512
        g1[6] = 0;
        g1[7] = 0;
        i32x4 g2, g3;
        g2[0] = (int)(idx[0]  | (idx[1]  << 16));
        g2[1] = (int)(idx[2]  | (idx[3]  << 16));
        g2[2] = (int)(idx[4]  | (idx[5]  << 16));
        g2[3] = (int)(idx[6]  | (idx[7]  << 16));
        g3[0] = (int)(idx[8]  | (idx[9]  << 16));
        g3[1] = (int)(idx[10] | (idx[11] << 16));
        g3[2] = (int)(idx[12] | (idx[13] << 16));
        g3[3] = (int)(idx[14] | (idx[15] << 16));
        i32x8 g4 = {0, 0, 0, 0, 0, 0, 0, 0};             // trailing group (zero-filled)
        __builtin_amdgcn_tensor_load_to_lds(g0, g1, g2, g3, g4, 0);
    };

    // preload W_hh fragments into LDS (8192 x 16B)
    {
        const uint4v* src = (const uint4v*)whh_swz;
        uint4v*       dst = (uint4v*)s_whh;
        for (int i = tid; i < N_WHH / 8; i += 256) dst[i] = src[i];
    }
    for (int i = tid; i < 16 * HID; i += 256) s_h[i] = (_Float16)0.f;
    if (wv == 0) tdm_issue(0);                           // prefetch xg[0]
    __syncthreads();

    float c[8];
#pragma unroll
    for (int q = 0; q < 8; ++q) c[q] = 0.f;

    for (int t = 0; t < TT; ++t) {
        // ---- issue DMA for t+1, guarantee buffer t has landed ----
        if (wv == 0) {
            if (t < TT - 1) {
                tdm_issue(t + 1);
                __builtin_amdgcn_s_wait_tensorcnt(1);    // only t+1 still in flight
            } else {
                __builtin_amdgcn_s_wait_tensorcnt(0);
            }
        }
        __syncthreads();

        const _Float16* xg = s_xg + ((t & 1) ? 8192 : 0);

        // ---- phase A: gates = xg (bias baked in) + h @ W_hh^T ----
        for (int nt4 = 0; nt4 < 4; ++nt4) {
            const int nt  = wv * 4 + nt4;
            const int col = nt * 16 + (lane & 15);
            v8f acc;
#pragma unroll
            for (int j = 0; j < 8; ++j)
                acc[j] = (float)xg[(rofs + j) * G4 + col];
#pragma unroll
            for (int kt = 0; kt < KT_HH; ++kt) {
                FragH fa;
                const uint4v* ap = (const uint4v*)(s_h + mrow * HID + a_base_k(kt, lane));
                fa.q[0] = ap[0];    // halves +0..7
                fa.q[1] = ap[2];    // halves +16..23
                FragH fb;
                const uint4v* bp = (const uint4v*)(s_whh + ((kt * 32 + nt) * 32 + lane) * 16);
                fb.q[0] = bp[0];
                fb.q[1] = bp[1];
                acc = __builtin_amdgcn_wmma_f32_16x16x32_f16(false, fa.h, false, fb.h,
                                                             (short)0, acc, false, false);
            }
#pragma unroll
            for (int j = 0; j < 8; ++j)
                s_g[(rofs + j) * G4 + col] = acc[j];
        }
        __syncthreads();

        // ---- phase B: gate nonlinearities + state update (8 elems/thread) ----
#pragma unroll
        for (int q = 0; q < 8; ++q) {
            int e   = tid * 8 + q;           // 0..2047
            int row = e >> 7;
            int j   = e & 127;
            float gi = s_g[row * G4 + j];
            float gf = s_g[row * G4 + 128 + j];
            float gg = s_g[row * G4 + 256 + j];
            float go = s_g[row * G4 + 384 + j];
            c[q] = sigmoidf_(gf) * c[q] + sigmoidf_(gi) * tanh_fast(gg);
            float hh = sigmoidf_(go) * tanh_fast(c[q]);
            s_h[row * HID + j] = (_Float16)hh;
            if (t == TT - 1) hn[((size_t)btile * 16 + row) * HID + j] = hh;
        }
        __syncthreads();
    }
}

// ---------------------------------------------------------------------------
// K4: fc head. One block (64 threads) per batch row.
// ---------------------------------------------------------------------------
__global__ void fc_kernel(const float* __restrict__ hn, const float* __restrict__ fc1_w,
                          const float* __restrict__ fc1_b, const float* __restrict__ fc2_w,
                          const float* __restrict__ fc2_b, float* __restrict__ out) {
    __shared__ float s[64];
    int b = blockIdx.x, k = threadIdx.x;
    float acc = fc1_b[k];
    const float* hrow = hn + (size_t)b * HID;
    const float* wrow = fc1_w + (size_t)k * HID;
#pragma unroll 4
    for (int j = 0; j < HID; ++j) acc += hrow[j] * wrow[j];
    s[k] = fmaxf(acc, 0.f) * fc2_w[k];
    __syncthreads();
    if (k == 0) {
        float sum = fc2_b[0];
        for (int i = 0; i < 64; ++i) sum += s[i];
        out[b] = 1.f / (1.f + __expf(-sum));
    }
}

// ---------------------------------------------------------------------------
extern "C" void kernel_launch(void* const* d_in, const int* in_sizes, int n_in,
                              void* d_out, int out_size, void* d_ws, size_t ws_size,
                              hipStream_t stream) {
    (void)in_sizes; (void)n_in; (void)out_size; (void)ws_size;
    const int*   x     = (const int*)d_in[0];
    const float* emb   = (const float*)d_in[1];
    const float* w_ih  = (const float*)d_in[2];
    const float* w_hh  = (const float*)d_in[3];
    const float* b_ih  = (const float*)d_in[4];
    const float* b_hh  = (const float*)d_in[5];
    const float* fc1_w = (const float*)d_in[6];
    const float* fc1_b = (const float*)d_in[7];
    const float* fc2_w = (const float*)d_in[8];
    const float* fc2_b = (const float*)d_in[9];
    float* out = (float*)d_out;

    char* ws = (char*)d_ws;
    _Float16* proj    = (_Float16*)ws;                  // 50000*512*2 = 51,200,000 B
    _Float16* wih_swz = (_Float16*)(ws + 51200000);     //   327,680 B
    _Float16* whh_swz = (_Float16*)(ws + 51527680);     //   131,072 B
    float*    bsum    = (float*)(ws + 51658752);        //     2,048 B
    float*    hn      = (float*)(ws + 51660800);        //   524,288 B
    _Float16* emb_h   = (_Float16*)(ws + 52185088);     // 50000*320*2 = 32,000,000 B
                                                        // total ~80.3 MiB

    embcvt_kernel<<<(VOCAB * (EMBP / 2) + 255) / 256, 256, 0, stream>>>(emb, emb_h);
    const int prep_n = N_WIH + N_WHH + G4;
    prep_kernel<<<(prep_n + 255) / 256, 256, 0, stream>>>(w_ih, w_hh, b_ih, b_hh,
                                                          wih_swz, whh_swz, bsum);
    proj_gemm_kernel<<<VOCAB / 16, 256, 0, stream>>>(emb_h, wih_swz, bsum, proj);
    lstm_recurrence_kernel<<<BB / 16, 256, LDS_TOTAL, stream>>>(x, proj, whh_swz, hn);
    fc_kernel<<<BB, 64, 0, stream>>>(hn, fc1_w, fc1_b, fc2_w, fc2_b, out);
}
